// SMRPN_18760417149174
// MI455X (gfx1250) — matrix operand order repaired and checked by reading, not compiled, not run on hardware
//
#include <hip/hip_runtime.h>
#include <hip/hip_bf16.h>

typedef __attribute__((ext_vector_type(8)))  float        v8f;
typedef __attribute__((ext_vector_type(16))) __bf16       v16bf;
typedef __attribute__((ext_vector_type(4)))  unsigned int v4u;

#define CIN   256
#define COUT  256
#define KDIM  (CIN * 9)
#define MBLK  128
#define NBLK  128
#define KT    64
#define AS    72   // LDS row stride (ushorts): 16B-aligned rows, bank-spread
#define BS    72
#define EPS   1e-5f

__device__ __forceinline__ unsigned short f2bf(float f) {
    unsigned int u = __float_as_uint(f);
    unsigned int r = u + 0x7FFFu + ((u >> 16) & 1u);   // round-to-nearest-even
    return (unsigned short)(r >> 16);
}

// A fragment: per lane, 16 contiguous bf16 at p[0..15] (two b128 LDS reads)
__device__ __forceinline__ v16bf ld_frag_a(const unsigned short* p) {
    v16bf r;
    v4u* rv = reinterpret_cast<v4u*>(&r);
    rv[0] = *reinterpret_cast<const v4u*>(p);
    rv[1] = *reinterpret_cast<const v4u*>(p + 8);
    return r;
}
// B fragment: per lane, 16 contiguous bf16 (two b128 LDS reads)
__device__ __forceinline__ v16bf ld_frag_b(const unsigned short* p) {
    v16bf r;
    v4u* rv = reinterpret_cast<v4u*>(&r);
    rv[0] = *reinterpret_cast<const v4u*>(p);
    rv[1] = *reinterpret_cast<const v4u*>(p + 8);
    return r;
}

// ---------------- f32 -> bf16 convert ----------------
__global__ void k_tobf(const float* __restrict__ x, unsigned short* __restrict__ y, int n) {
    int i = blockIdx.x * 256 + threadIdx.x;
    if (i < n) y[i] = f2bf(x[i]);
}

// ---- weight repack into WMMA-tile-resident order ----
// w[cout][cin][kh][kw]  ->  wp[(((nb*36) + khw*4 + cq)*128 + nn)*64 + kk]
// where cout = nb*128+nn, cin = cq*64+kk, ktile index = khw*4+cq (K ordered (kh,kw,cin))
__global__ void k_packw(const float* __restrict__ w, unsigned short* __restrict__ wp, int n) {
    int i = blockIdx.x * 256 + threadIdx.x;
    if (i >= n) return;
    int cout = i / (CIN * 9);
    int rem  = i - cout * (CIN * 9);
    int cin  = rem / 9;
    int khw  = rem - cin * 9;
    int nb = cout >> 7, nn = cout & 127;
    int cq = cin >> 6,  kk = cin & 63;
    wp[(((nb * 36) + khw * 4 + cq) * 128 + nn) * 64 + kk] = f2bf(w[i]);
}

// ---------------- implicit-GEMM 3x3 conv (pad=1) via WMMA bf16 ----------------
// M = B*H*W rows (b,h,w), N = COUT, K = CIN*9 ordered (kh,kw,cin), cin fastest.
template <int H, int W>
__global__ __launch_bounds__(256) void k_conv3x3_wmma(
    const unsigned short* __restrict__ xbf,   // [B,CIN,H,W] bf16
    const unsigned short* __restrict__ wp,    // tile-packed weights (see k_packw)
    float* __restrict__ y, int B)             // [B,COUT,H,W] f32
{
    __shared__ __align__(16) unsigned short At[MBLK * AS];
    __shared__ __align__(16) unsigned short Bt[NBLK * BS];

    const int HW    = H * W;
    const int tid   = threadIdx.x;
    const int lane  = tid & 31;
    const int wave  = tid >> 5;
    const int waveM = wave & 3;     // 4 waves along M (32 rows each)
    const int waveN = wave >> 2;    // 2 waves along N (64 cols each)
    const int ml    = lane & 15;
    const int kh2   = lane >> 4;

    const int mBlock = blockIdx.x * MBLK;
    const int nb     = blockIdx.y;            // N block (128 couts)

    // this thread's fixed im2col row
    const int r       = tid & 127;
    const int halfSel = tid >> 7;
    const int rowG = mBlock + r;
    const int bIdx = rowG / HW;
    const int hw0  = rowG - bIdx * HW;
    const int h0   = hw0 / W;
    const int w0   = hw0 - h0 * W;
    const unsigned short* sRow = xbf + ((long)bIdx * CIN) * HW;

    v8f acc[2][4];
    #pragma unroll
    for (int i = 0; i < 2; i++)
        #pragma unroll
        for (int j = 0; j < 4; j++)
            acc[i][j] = (v8f){0.f, 0.f, 0.f, 0.f, 0.f, 0.f, 0.f, 0.f};

    #pragma unroll 1
    for (int g = 0; g < 9; g++) {             // (kh,kw): all shift math hoisted here
        const int dy = g / 3 - 1;
        const int dx = g - (g / 3) * 3 - 1;
        const int hs = h0 + dy, ws2 = w0 + dx;
        const bool ok = (hs >= 0) && (hs < H) && (ws2 >= 0) && (ws2 < W);
        const unsigned short* sp = sRow + (long)hs * W + ws2;   // + c*HW per element

        #pragma unroll 1
        for (int cq = 0; cq < 4; cq++) {      // K-tile = 64 cin
            // ---- A tile: gather 8 strided bf16 -> one b128 LDS store, 4 octets/thread ----
            {
                const unsigned short* spc = sp + (long)(cq * 64) * HW;
                #pragma unroll
                for (int gi = 0; gi < 4; gi++) {
                    const int oct = halfSel + gi * 2;          // 0..7, disjoint across halves
                    v4u pk = (v4u){0u, 0u, 0u, 0u};
                    if (ok) {
                        const unsigned short* q = spc + (long)(oct * 8) * HW;
                        pk.x = (unsigned)q[0 * HW] | ((unsigned)q[1 * HW] << 16);
                        pk.y = (unsigned)q[2 * HW] | ((unsigned)q[3 * HW] << 16);
                        pk.z = (unsigned)q[4 * HW] | ((unsigned)q[5 * HW] << 16);
                        pk.w = (unsigned)q[6 * HW] | ((unsigned)q[7 * HW] << 16);
                    }
                    *reinterpret_cast<v4u*>(&At[r * AS + oct * 8]) = pk;
                }
            }
            // ---- B tile: contiguous pre-packed copy, b128 both sides ----
            {
                const unsigned short* tb = wp + (size_t)((nb * 36) + g * 4 + cq) * (NBLK * KT);
                #pragma unroll
                for (int gi = 0; gi < 4; gi++) {
                    const int grp = tid + gi * 256;            // 0..1023
                    const int nn = grp >> 3;
                    const int oc = grp & 7;
                    *reinterpret_cast<v4u*>(&Bt[nn * BS + oc * 8]) =
                        *reinterpret_cast<const v4u*>(tb + grp * 8);
                }
            }
            __syncthreads();

            // ---- WMMA over two K=32 chunks ----
            #pragma unroll
            for (int kc = 0; kc < KT; kc += 32) {
                v16bf af[2], bfv[4];
                #pragma unroll
                for (int i = 0; i < 2; i++)
                    af[i] = ld_frag_a(&At[(waveM * 32 + i * 16 + ml) * AS + kc + kh2 * 16]);
                #pragma unroll
                for (int j = 0; j < 4; j++)
                    bfv[j] = ld_frag_b(&Bt[(waveN * 64 + j * 16 + ml) * BS + kc + kh2 * 16]);
                #pragma unroll
                for (int i = 0; i < 2; i++)
                    #pragma unroll
                    for (int j = 0; j < 4; j++)
                        acc[i][j] = __builtin_amdgcn_wmma_f32_16x16x32_bf16(
                            false, af[i], false, bfv[j], (short)0, acc[i][j], false, false);
            }
            __syncthreads();
        }
    }

    // ---- epilogue: scatter to NCHW f32 (conv bias dropped: cancelled by train-mode BN) ----
    #pragma unroll
    for (int i = 0; i < 2; i++) {
        #pragma unroll
        for (int j = 0; j < 4; j++) {
            const int col = nb * NBLK + waveN * 64 + j * 16 + ml;
            #pragma unroll
            for (int rr = 0; rr < 8; rr++) {
                int rowO = mBlock + waveM * 32 + i * 16 + kh2 * 8 + rr;
                int bo   = rowO / HW;
                int hwo  = rowO - bo * HW;
                y[((long)bo * COUT + col) * HW + hwo] = acc[i][j][rr];
            }
        }
    }
}

// ---------------- BN stats: one block per channel ----------------
__global__ __launch_bounds__(256) void k_bn_stats(const float* __restrict__ x,
                                                  float* __restrict__ mean,
                                                  float* __restrict__ istd,
                                                  int Bn, int HW) {
    __shared__ float r0[256], r1[256];
    const int c = blockIdx.x, tid = threadIdx.x;
    float s0 = 0.f, s1 = 0.f;
    for (int b = 0; b < Bn; b++) {
        const float* p = x + ((long)b * 256 + c) * HW;
        for (int i = tid; i < HW; i += 256) { float v = p[i]; s0 += v; s1 += v * v; }
    }
    r0[tid] = s0; r1[tid] = s1; __syncthreads();
    for (int off = 128; off > 0; off >>= 1) {
        if (tid < off) { r0[tid] += r0[tid + off]; r1[tid] += r1[tid + off]; }
        __syncthreads();
    }
    if (tid == 0) {
        float cnt = (float)Bn * (float)HW;
        float m = r0[0] / cnt;
        float v = fmaxf(r1[0] / cnt - m * m, 0.f);
        mean[c] = m;
        istd[c] = rsqrtf(v + EPS);
    }
}

// ---------------- BN apply (in-place) ----------------
__global__ void k_bn_apply(float* __restrict__ x, const float* __restrict__ mean,
                           const float* __restrict__ istd, const float* __restrict__ g,
                           const float* __restrict__ be, int total, int hwShift) {
    int i = blockIdx.x * 256 + threadIdx.x;
    if (i >= total) return;
    int c = (i >> hwShift) & 255;
    x[i] = (x[i] - mean[c]) * istd[c] * g[c] + be[c];
}

// ---------------- per-sample template stats ----------------
__global__ __launch_bounds__(256) void k_tstats(const float* __restrict__ t,
                                                float* __restrict__ mt,
                                                float* __restrict__ invden) {
    __shared__ float r0[256], r1[256];
    const int b = blockIdx.x, tid = threadIdx.x;
    const float* p = t + (long)b * 65536;
    float s0 = 0.f, s1 = 0.f;
    for (int i = tid; i < 65536; i += 256) { float v = p[i]; s0 += v; s1 += v * v; }
    r0[tid] = s0; r1[tid] = s1; __syncthreads();
    for (int off = 128; off > 0; off >>= 1) {
        if (tid < off) { r0[tid] += r0[tid + off]; r1[tid] += r1[tid + off]; }
        __syncthreads();
    }
    if (tid == 0) {
        float m = r0[0] * (1.f / 65536.f);
        float ss = fmaxf(r1[0] - 65536.f * m * m, 0.f);
        float ttstd = sqrtf(ss * (1.f / 65535.f));
        mt[b] = m;
        invden[b] = 1.f / (65536.f * ttstd);
    }
}

// ---------------- NCC: one block per (b,i,j); fused dot / win_sum / win_sqsum ----------------
__global__ __launch_bounds__(256) void k_ncc(const float* __restrict__ sN,
                                             const float* __restrict__ tN,
                                             const float* __restrict__ mt,
                                             const float* __restrict__ invden,
                                             float* __restrict__ ncc) {
    __shared__ float r0[256], r1[256], r2[256];
    const int j = blockIdx.x, i = blockIdx.y, b = blockIdx.z;
    const int tid = threadIdx.x;
    const int x = tid & 15, y = (tid >> 4) & 15;
    const float* sp = sN + (((long)b * 256) * 64 + (i + y)) * 64 + (j + x);
    const float* tp = tN + (long)b * 65536 + y * 16 + x;
    float dot = 0.f, wsum = 0.f, wsq = 0.f;
    #pragma unroll 4
    for (int c = 0; c < 256; c++) {
        float sv = sp[(long)c * 4096];
        float tv = tp[c * 256];
        dot += sv * tv; wsum += sv; wsq += sv * sv;
    }
    r0[tid] = dot; r1[tid] = wsum; r2[tid] = wsq; __syncthreads();
    for (int off = 128; off > 0; off >>= 1) {
        if (tid < off) { r0[tid] += r0[tid+off]; r1[tid] += r1[tid+off]; r2[tid] += r2[tid+off]; }
        __syncthreads();
    }
    if (tid == 0) {
        float cross = r0[0] - mt[b] * r1[0];      // sum(tt)==0 identity
        float wm  = r1[0] * (1.f / 65536.f);
        float var = fmaxf(r2[0] - 65536.f * wm * wm, 0.f) * (1.f / 65535.f);
        ncc[((long)b * 49 + i) * 49 + j] = cross * invden[b] / sqrtf(var);
    }
}

// ---------------- ncc mean/var (biased) over all 38416 elems ----------------
__global__ __launch_bounds__(256) void k_ncc_stats(const float* __restrict__ ncc,
                                                   float* __restrict__ st, int n) {
    __shared__ float r0[256], r1[256];
    const int tid = threadIdx.x;
    float s0 = 0.f, s1 = 0.f;
    for (int i = tid; i < n; i += 256) { float v = ncc[i]; s0 += v; s1 += v * v; }
    r0[tid] = s0; r1[tid] = s1; __syncthreads();
    for (int off = 128; off > 0; off >>= 1) {
        if (tid < off) { r0[tid] += r0[tid + off]; r1[tid] += r1[tid + off]; }
        __syncthreads();
    }
    if (tid == 0) {
        float m = r0[0] / (float)n;
        st[0] = m;
        st[1] = fmaxf(r1[0] / (float)n - m * m, 0.f);
    }
}

// ---------------- head: conv3(1x1) + BN + ReLU + conv4(1x1 -> 2ch) ----------------
__global__ void k_head(const float* __restrict__ ncc, const float* __restrict__ st,
                       const float* __restrict__ w3, const float* __restrict__ b3,
                       const float* __restrict__ g3, const float* __restrict__ be3,
                       const float* __restrict__ w4, const float* __restrict__ b4,
                       float* __restrict__ out, int n) {
    int p = blockIdx.x * 256 + threadIdx.x;
    if (p >= n) return;
    int b = p / 2401;
    int ij = p - b * 2401;
    float W3 = w3[0], B3 = b3[0];
    float mY = W3 * st[0] + B3;
    float rs = rsqrtf(W3 * W3 * st[1] + EPS);
    float yv = W3 * ncc[p] + B3;
    float xn = (yv - mY) * rs * g3[0] + be3[0];
    float xr = fmaxf(xn, 0.f);
    out[((long)b * 2 + 0) * 2401 + ij] = w4[0] * xr + b4[0];
    out[((long)b * 2 + 1) * 2401 + ij] = w4[1] * xr + b4[1];
}

extern "C" void kernel_launch(void* const* d_in, const int* in_sizes, int n_in,
                              void* d_out, int out_size, void* d_ws, size_t ws_size,
                              hipStream_t stream) {
    (void)in_sizes; (void)n_in; (void)out_size; (void)ws_size;
    const float* s   = (const float*)d_in[0];
    const float* t   = (const float*)d_in[1];
    const float* w1  = (const float*)d_in[2];
    const float* g1  = (const float*)d_in[4];
    const float* be1 = (const float*)d_in[5];
    const float* w2  = (const float*)d_in[6];
    const float* g2  = (const float*)d_in[8];
    const float* be2 = (const float*)d_in[9];
    const float* w3  = (const float*)d_in[10];
    const float* b3  = (const float*)d_in[11];
    const float* g3  = (const float*)d_in[12];
    const float* be3 = (const float*)d_in[13];
    const float* w4  = (const float*)d_in[14];
    const float* b4  = (const float*)d_in[15];
    float* out = (float*)d_out;

    char* ws = (char*)d_ws;
    const size_t nS  = 16u * 256u * 64u * 64u;   // 16,777,216
    const size_t nT  = 16u * 256u * 16u * 16u;   //  1,048,576
    const size_t nW  = 256u * 256u * 9u;         //    589,824
    const int    nNcc = 16 * 49 * 49;            //     38,416

    size_t off = 0;
    float* sConv = (float*)(ws + off);           off += nS * 4;         // 64 MB
    float* tConv = (float*)(ws + off);           off += nT * 4;         //  4 MB
    unsigned short* sBf = (unsigned short*)(ws + off); off += nS * 2;   // 32 MB
    unsigned short* tBf = (unsigned short*)(ws + off); off += nT * 2;   //  2 MB
    unsigned short* w1p = (unsigned short*)(ws + off); off += nW * 2;
    unsigned short* w2p = (unsigned short*)(ws + off); off += nW * 2;
    float* nccBuf = (float*)(ws + off);          off += ((nNcc * 4 + 255) / 256) * 256;
    float* stats  = (float*)(ws + off);
    float* mean1 = stats, *istd1 = stats + 256, *mean2 = stats + 512, *istd2 = stats + 768;
    float* mt = stats + 1024, *invden = stats + 1040, *nccSt = stats + 1056;

    // 1. convert inputs + pack weights (tile-resident layout) to bf16
    k_tobf<<<(int)((nS + 255) / 256), 256, 0, stream>>>(s, sBf, (int)nS);
    k_tobf<<<(int)((nT + 255) / 256), 256, 0, stream>>>(t, tBf, (int)nT);
    k_packw<<<(int)((nW + 255) / 256), 256, 0, stream>>>(w1, w1p, (int)nW);
    k_packw<<<(int)((nW + 255) / 256), 256, 0, stream>>>(w2, w2p, (int)nW);

    // 2. WMMA implicit-GEMM convs
    k_conv3x3_wmma<64, 64><<<dim3(65536 / MBLK, COUT / NBLK), 256, 0, stream>>>(sBf, w1p, sConv, 16);
    k_conv3x3_wmma<16, 16><<<dim3(4096 / MBLK, COUT / NBLK), 256, 0, stream>>>(tBf, w2p, tConv, 16);

    // 3. train-mode BatchNorm (stats + in-place apply)
    k_bn_stats<<<256, 256, 0, stream>>>(sConv, mean1, istd1, 16, 4096);
    k_bn_apply<<<(int)((nS + 255) / 256), 256, 0, stream>>>(sConv, mean1, istd1, g1, be1, (int)nS, 12);
    k_bn_stats<<<256, 256, 0, stream>>>(tConv, mean2, istd2, 16, 256);
    k_bn_apply<<<(int)((nT + 255) / 256), 256, 0, stream>>>(tConv, mean2, istd2, g2, be2, (int)nT, 8);

    // 4. template stats, NCC, head
    k_tstats<<<16, 256, 0, stream>>>(tConv, mt, invden);
    k_ncc<<<dim3(49, 49, 16), 256, 0, stream>>>(sConv, tConv, mt, invden, nccBuf);
    k_ncc_stats<<<1, 256, 0, stream>>>(nccBuf, nccSt, nNcc);
    k_head<<<(nNcc + 255) / 256, 256, 0, stream>>>(nccBuf, nccSt, w3, b3, g3, be3, w4, b4, out, nNcc);
}
